// CustomTripletLoss_23570780520583
// MI455X (gfx1250) — matrix operand order repaired and checked by reading, not compiled, not run on hardware
//
#include <hip/hip_runtime.h>
#include <stdint.h>

// ---- CDNA5 WMMA vector types ----
typedef __attribute__((ext_vector_type(16))) __bf16 v16bf;
typedef __attribute__((ext_vector_type(8)))  __bf16 v8bf;
typedef __attribute__((ext_vector_type(4)))  __bf16 v4bf;
typedef __attribute__((ext_vector_type(8)))  float  v8f;

#define B_N    16384
#define C_N    1000
#define D_N    1024
#define C_PAD  1008          // 63 * 16 column tiles
#define M_BLK  32            // anchor rows per workgroup (2 x 16-row WMMA tiles)
#define MARGIN_F 1.0f
#define EPS_F    1e-6f
#define A_STRIDE 1032        // 1024 + 8 bf16 pad -> row stride 2064B (16B aligned, bank-skewed)

// ---------------------------------------------------------------------------
// Kernel 1: per-target-row squared norm (fp32) + fp32 -> bf16 conversion.
// Padded rows (c >= C_N) get zero data and t2 = +inf so they never win argmin.
// ---------------------------------------------------------------------------
__global__ __launch_bounds__(256)
void prep_target(const float* __restrict__ target,
                 __bf16* __restrict__ tbf,
                 float* __restrict__ t2) {
    const int c   = blockIdx.x;
    const int tid = threadIdx.x;
    __shared__ float red[256];

    float s = 0.0f;
    if (c < C_N) {
        const float4 v = *(reinterpret_cast<const float4*>(target + (size_t)c * D_N) + tid);
        s = v.x * v.x + v.y * v.y + v.z * v.z + v.w * v.w;
        v4bf o = { (__bf16)v.x, (__bf16)v.y, (__bf16)v.z, (__bf16)v.w };
        *(reinterpret_cast<v4bf*>(tbf + (size_t)c * D_N) + tid) = o;
    } else {
        v4bf z = {};
        *(reinterpret_cast<v4bf*>(tbf + (size_t)c * D_N) + tid) = z;
    }
    red[tid] = s;
    __syncthreads();
    for (int off = 128; off > 0; off >>= 1) {
        if (tid < off) red[tid] += red[tid + off];
        __syncthreads();
    }
    if (tid == 0) t2[c] = (c < C_N) ? red[0] : __builtin_inff();
}

// ---------------------------------------------------------------------------
// Kernel 2: negative mining via bf16 WMMA.
// Block = 32 anchor rows (two 16-row tiles in LDS), 4 waves; each wave sweeps
// 16-column tiles of target, reusing each B fragment for BOTH row tiles
// (2 accumulators) -> 32 FLOP per L2 byte of B traffic.
// Score minimized: t2[n] - 2 * <x_m, t_n>   (x^2 term drops from argmin).
// ---------------------------------------------------------------------------
__global__ __launch_bounds__(128)
void mine(const float* __restrict__ inputs,
          const long long* __restrict__ labels,
          const __bf16* __restrict__ tbf,
          const float* __restrict__ t2,
          int* __restrict__ neg_idx) {
    __shared__ __bf16 Atile[M_BLK][A_STRIDE];
    __shared__ int    slab[M_BLK];
    __shared__ float  redv[4][M_BLK];
    __shared__ int    redi[4][M_BLK];

    const int tid     = threadIdx.x;
    const int wave    = tid >> 5;
    const int lane    = tid & 31;
    const int half    = lane >> 4;
    const int l15     = lane & 15;
    const int rowBase = blockIdx.x * M_BLK;

    // Stage A tiles (32 rows x 1024) fp32 -> bf16 into LDS.
    for (int i = tid; i < M_BLK * 256; i += 128) {
        const int r  = i >> 8;
        const int kq = i & 255;
        const float4 v = *(reinterpret_cast<const float4*>(inputs + (size_t)(rowBase + r) * D_N) + kq);
        v4bf o = { (__bf16)v.x, (__bf16)v.y, (__bf16)v.z, (__bf16)v.w };
        *reinterpret_cast<v4bf*>(&Atile[r][kq * 4]) = o;
    }
    if (tid < M_BLK) slab[tid] = (int)labels[rowBase + tid];
    __syncthreads();

    int myLab[16];
#pragma unroll
    for (int g = 0; g < 8; ++g) {
        myLab[g]     = slab[g + 8 * half];        // rows of tile 0
        myLab[8 + g] = slab[16 + g + 8 * half];   // rows of tile 1
    }

    float minv[16];
    int   mini[16];
#pragma unroll
    for (int j = 0; j < 16; ++j) { minv[j] = __builtin_inff(); mini[j] = 0; }

    // Each wave handles column tiles wave, wave+4, ...
    for (int t = wave; t < C_PAD / 16; t += 4) {
        const int col    = t * 16 + l15;          // this lane's N column
        const __bf16* tb = tbf + (size_t)col * D_N;

        v8f acc0 = {};
        v8f acc1 = {};
        for (int k0 = 0; k0 < D_N; k0 += 32) {
            // B fragment (32x16 bf16): lane = col l15; chunks at k0+8*half and k0+16+8*half
            v8bf blo = *reinterpret_cast<const v8bf*>(tb + k0 + 8 * half);
            v8bf bhi = *reinterpret_cast<const v8bf*>(tb + k0 + 16 + 8 * half);
            v16bf b = __builtin_shufflevector(blo, bhi,
                        0,1,2,3,4,5,6,7,8,9,10,11,12,13,14,15);

            // A fragments (16x32 bf16): lane = row l15 (+16 for tile 1);
            // chunks at k0+8*half and k0+16+8*half (ds_load_b128 from LDS)
            v8bf a0lo = *reinterpret_cast<const v8bf*>(&Atile[l15][k0 + 8 * half]);
            v8bf a0hi = *reinterpret_cast<const v8bf*>(&Atile[l15][k0 + 16 + 8 * half]);
            v16bf a0 = __builtin_shufflevector(a0lo, a0hi,
                        0,1,2,3,4,5,6,7,8,9,10,11,12,13,14,15);
            v8bf a1lo = *reinterpret_cast<const v8bf*>(&Atile[16 + l15][k0 + 8 * half]);
            v8bf a1hi = *reinterpret_cast<const v8bf*>(&Atile[16 + l15][k0 + 16 + 8 * half]);
            v16bf a1 = __builtin_shufflevector(a1lo, a1hi,
                        0,1,2,3,4,5,6,7,8,9,10,11,12,13,14,15);

            acc0 = __builtin_amdgcn_wmma_f32_16x16x32_bf16(
                       false, a0, false, b, (short)0, acc0, false, false);
            acc1 = __builtin_amdgcn_wmma_f32_16x16x32_bf16(
                       false, a1, false, b, (short)0, acc1, false, false);
        }

        const float t2c = t2[col];
#pragma unroll
        for (int g = 0; g < 8; ++g) {
            // acc0[g]: row (rowBase + g + 8*half);  acc1[g]: row (rowBase + 16 + g + 8*half)
            float v0 = t2c - 2.0f * acc0[g];
            float v1 = t2c - 2.0f * acc1[g];
            if (myLab[g]     == col) v0 = __builtin_inff();
            if (myLab[8 + g] == col) v1 = __builtin_inff();
            if (v0 < minv[g])     { minv[g]     = v0; mini[g]     = col; }
            if (v1 < minv[8 + g]) { minv[8 + g] = v1; mini[8 + g] = col; }
        }
    }

    // Reduce (min, idx) across the 16 lanes holding the same row.
#pragma unroll
    for (int j = 0; j < 16; ++j) {
        float v = minv[j];
        int   i = mini[j];
        for (int m = 8; m > 0; m >>= 1) {
            float ov = __shfl_xor(v, m, 16);
            int   oi = __shfl_xor(i, m, 16);
            if (ov < v || (ov == v && oi < i)) { v = ov; i = oi; }
        }
        if (l15 == 0) {
            const int g   = j & 7;
            const int pos = (j < 8) ? (g + 8 * half) : (16 + g + 8 * half);
            redv[wave][pos] = v;
            redi[wave][pos] = i;
        }
    }
    __syncthreads();

    if (tid < M_BLK) {
        float v = redv[0][tid];
        int   i = redi[0][tid];
        for (int w = 1; w < 4; ++w) {
            float ov = redv[w][tid];
            int   oi = redi[w][tid];
            if (ov < v || (ov == v && oi < i)) { v = ov; i = oi; }
        }
        neg_idx[rowBase + tid] = i;
    }
}

// ---------------------------------------------------------------------------
// Kernel 3: exact fp32 pairwise distances + hinge, one block per anchor.
// ---------------------------------------------------------------------------
__global__ __launch_bounds__(256)
void finalize(const float* __restrict__ inputs,
              const long long* __restrict__ labels,
              const float* __restrict__ target,
              const int* __restrict__ neg_idx,
              float* __restrict__ per) {
    const int r   = blockIdx.x;
    const int tid = threadIdx.x;
    __shared__ float rap[256];
    __shared__ float ran[256];

    const float4* x  = reinterpret_cast<const float4*>(inputs + (size_t)r * D_N);
    const float4* tp = reinterpret_cast<const float4*>(target + (size_t)labels[r]  * D_N);
    const float4* tn = reinterpret_cast<const float4*>(target + (size_t)neg_idx[r] * D_N);

    const float4 xv = x[tid];
    const float4 pv = tp[tid];
    const float4 nv = tn[tid];

    float a0 = xv.x - pv.x + EPS_F, a1 = xv.y - pv.y + EPS_F;
    float a2 = xv.z - pv.z + EPS_F, a3 = xv.w - pv.w + EPS_F;
    float b0 = xv.x - nv.x + EPS_F, b1 = xv.y - nv.y + EPS_F;
    float b2 = xv.z - nv.z + EPS_F, b3 = xv.w - nv.w + EPS_F;

    rap[tid] = a0*a0 + a1*a1 + a2*a2 + a3*a3;
    ran[tid] = b0*b0 + b1*b1 + b2*b2 + b3*b3;
    __syncthreads();
    for (int off = 128; off > 0; off >>= 1) {
        if (tid < off) { rap[tid] += rap[tid + off]; ran[tid] += ran[tid + off]; }
        __syncthreads();
    }
    if (tid == 0) {
        const float d = sqrtf(rap[0]) - sqrtf(ran[0]) + MARGIN_F;
        per[r] = d > 0.0f ? d : 0.0f;
    }
}

// ---------------------------------------------------------------------------
// Kernel 4: deterministic mean over B anchors (fixed summation order).
// ---------------------------------------------------------------------------
__global__ __launch_bounds__(256)
void mean_kernel(const float* __restrict__ per, float* __restrict__ out) {
    const int tid = threadIdx.x;
    __shared__ float red[256];
    float s = 0.0f;
    for (int i = tid; i < B_N; i += 256) s += per[i];
    red[tid] = s;
    __syncthreads();
    for (int off = 128; off > 0; off >>= 1) {
        if (tid < off) red[tid] += red[tid + off];
        __syncthreads();
    }
    if (tid == 0) out[0] = red[0] * (1.0f / (float)B_N);
}

// ---------------------------------------------------------------------------
extern "C" void kernel_launch(void* const* d_in, const int* in_sizes, int n_in,
                              void* d_out, int out_size, void* d_ws, size_t ws_size,
                              hipStream_t stream) {
    (void)in_sizes; (void)n_in; (void)out_size; (void)ws_size;

    const float*     inputs = (const float*)d_in[0];
    const long long* labels = (const long long*)d_in[1];  // int64 labels
    const float*     target = (const float*)d_in[2];

    char* ws = (char*)d_ws;
    size_t off = 0;
    __bf16* tbf = (__bf16*)(ws + off);  off += (size_t)C_PAD * D_N * sizeof(__bf16); // 2,064,384
    float*  t2  = (float*)(ws + off);   off += 4096;                                  // C_PAD*4 padded
    int*    ngi = (int*)(ws + off);     off += (size_t)B_N * sizeof(int);
    float*  per = (float*)(ws + off);
    float*  out = (float*)d_out;

    prep_target<<<C_PAD, 256, 0, stream>>>(target, tbf, t2);
    mine<<<B_N / M_BLK, 128, 0, stream>>>(inputs, labels, tbf, t2, ngi);
    finalize<<<B_N, 256, 0, stream>>>(inputs, labels, target, ngi, per);
    mean_kernel<<<1, 256, 0, stream>>>(per, out);
}
